// SimpleLSTM_360777253155
// MI455X (gfx1250) — matrix operand order repaired and checked by reading, not compiled, not run on hardware
//
#include <hip/hip_runtime.h>

// ---------------------------------------------------------------------------
// sLSTM scan for MI455X (gfx1250).
//  * bf16 WMMA (v_wmma_f32_16x16x32_bf16), f32 accumulation.
//  * Weights + pre-packed x fragments live in L2 (13.6 MB + 33.5 MB << 192 MB).
//  * One fused kernel per timestep: gates(t) + out(t-1); h passed between
//    steps directly in WMMA A-fragment layout (no per-step conversion).
// ---------------------------------------------------------------------------

#define kB    32
#define kT    1024
#define kDin  512
#define kDh   1024
#define kDout 512
#define kKGate   (kDin + kDh)     // 1536
#define kKsGate  (kKGate / 32)    // 48 K-steps for gate GEMMs
#define kKsX     (kDin / 32)      // 16 K-steps covering x
#define kKsH     (kDh / 32)       // 32 K-steps covering h (also out GEMM)
#define kNtGate  (kDh / 16)       // 64 N-tiles (gates)
#define kNtOut   (kDout / 16)     // 32 N-tiles (out)
#define kGateStride   (kNtGate * kKsGate * 512)   // halfs per gate weight
#define kGateSwzHalfs (4 * kGateStride)           // 6291456
#define kOutSwzHalfs  (kNtOut * kKsH * 512)       // 524288
#define kXPackPerT    (kKsX * 2 * 512)            // 16384 halfs per timestep
#define kHPackHalfs   (kKsH * 2 * 512)            // 32768 halfs per h buffer

typedef __attribute__((ext_vector_type(16))) __bf16 v16bf;
typedef __attribute__((ext_vector_type(8)))  float  v8f;

__device__ __forceinline__ unsigned short f2bf(float f) {
  unsigned int u = __float_as_uint(f);
  unsigned int r = u + 0x7fffu + ((u >> 16) & 1u);   // round-to-nearest-even
  if ((u & 0x7f800000u) == 0x7f800000u) r = u;       // inf/nan: truncate
  return (unsigned short)(r >> 16);
}

// A-fragment addressing for 16-bit A 16x32 (ISA 7.12.2):
// lane 0-15: row M=lane, K={0..7} then {16..23}; lane 16-31: row M=lane-16,
// K={8..15} then {24..31}; 16 halfs per lane.
__device__ __forceinline__ int afrag_kk(int lane, int j) {
  int hi = lane >> 4;
  return (j < 8) ? (hi * 8 + j) : (16 + hi * 8 + (j - 8));
}

#define WMMA_BF16(A, Bm, C) \
  __builtin_amdgcn_wmma_f32_16x16x32_bf16(false, (A), false, (Bm), (short)0, (C), false, false)

// ---------------------------------------------------------------------------
__global__ void zero_f32(float* p, int count) {
  int i = blockIdx.x * 256 + threadIdx.x;
  if (i < count) p[i] = 0.0f;
}

// One-time: swizzle f32 row-major weights -> bf16 B-fragment tiles.
// B tile (32x16 bf16, 1KB): lane 0-15 hold K=0..7,16..23 of col N=lane; lane 16-31
// hold K=8..15,24..31 of col N=lane-16; 16 contiguous halfs per lane.
__global__ void swizzle_weights(const float* __restrict__ Wf, const float* __restrict__ Wi,
                                const float* __restrict__ Wc, const float* __restrict__ Wo,
                                const float* __restrict__ Wout,
                                unsigned short* __restrict__ ws_us) {
  int idx  = blockIdx.x * 256 + threadIdx.x;        // exact: 6815744 = 26624*256
  int lane = (idx >> 4) & 31;
  int j    = idx & 15;
  int kk   = afrag_kk(lane, j);
  int nl   = lane & 15;
  if (idx < kGateSwzHalfs) {
    int b      = idx >> 9;
    int k_step = b % kKsGate;
    int n_tile = (b / kKsGate) % kNtGate;
    int g      = b / (kKsGate * kNtGate);
    const float* W = (g == 0) ? Wf : (g == 1) ? Wi : (g == 2) ? Wc : Wo;
    ws_us[idx] = f2bf(W[(k_step * 32 + kk) * kDh + n_tile * 16 + nl]);
  } else {
    int i2     = idx - kGateSwzHalfs;
    int b      = i2 >> 9;
    int k_step = b % kKsH;
    int n_tile = b / kKsH;
    ws_us[idx] = f2bf(Wout[(k_step * 32 + kk) * kDout + n_tile * 16 + nl]);
  }
}

// One-time: pack all timesteps of x into bf16 A-fragment layout:
// xpack[t][ks 0..15][mt 0..1][lane][j].
__global__ void pack_x(const float* __restrict__ x, unsigned short* __restrict__ xpack) {
  int idx = blockIdx.x * 256 + threadIdx.x;         // exact: 16777216 = 65536*256
  int t    = idx >> 14;
  int rem  = idx & 16383;
  int frag = rem >> 9;                              // 0..31
  int ks = frag >> 1, mt = frag & 1;
  int lane = (rem >> 4) & 31, j = rem & 15;
  int K = ks * 32 + afrag_kk(lane, j);
  int M = mt * 16 + (lane & 15);
  xpack[idx] = f2bf(x[(M * kT + t) * kDin + K]);
}

// ---------------------------------------------------------------------------
// Fused per-step kernel: grid = 24 blocks x 128 threads (4 waves).
//   blocks  0..15 : gates(t)  -- 64 N-tile waves; reads xpack[t] + hp_rd,
//                   updates c/n, writes h f32 (d_out tail) and hp_wr frags.
//   blocks 16..23 : out(t-1)  -- 32 N-tile waves; reads hp_rd (h_{t-1} frags).
// Both roles only READ hp_rd; gates WRITES hp_wr => no intra-kernel hazard.
// ---------------------------------------------------------------------------
__global__ void lstm_step(const unsigned short* __restrict__ xpack,
                          const unsigned short* __restrict__ hp_rd,
                          unsigned short* __restrict__ hp_wr,
                          const unsigned short* __restrict__ wswz,
                          const unsigned short* __restrict__ wout_swz,
                          const float* __restrict__ bf_, const float* __restrict__ bi_,
                          const float* __restrict__ bc_, const float* __restrict__ bo_,
                          const float* __restrict__ bout,
                          float* __restrict__ h_tail, float* __restrict__ c,
                          float* __restrict__ nst, float* __restrict__ out, int t) {
  const int tid  = threadIdx.x;
  const int wave = tid >> 5;
  const int lane = tid & 31;
  const int laneoff = lane << 4;

  if (blockIdx.x < 16) {
    // ---------------- gates(t) ----------------
    const int n_tile = blockIdx.x * 4 + wave;       // 0..63
    const int ncol   = n_tile * 16 + (lane & 15);

    v8f aF0, aF1, aI0, aI1, aC0, aC1, aO0, aO1;
    {
      float vF = bf_[ncol], vI = bi_[ncol], vC = bc_[ncol], vO = bo_[ncol];
      for (int r = 0; r < 8; ++r) {
        aF0[r] = vF; aF1[r] = vF; aI0[r] = vI; aI1[r] = vI;
        aC0[r] = vC; aC1[r] = vC; aO0[r] = vO; aO1[r] = vO;
      }
    }

    const unsigned short* xa = xpack + t * kXPackPerT;
    for (int ks = 0; ks < kKsX; ++ks) {             // K 0..511 (x part of comb)
      v16bf A0 = *(const v16bf*)(xa + (ks * 2 + 0) * 512 + laneoff);
      v16bf A1 = *(const v16bf*)(xa + (ks * 2 + 1) * 512 + laneoff);
      const unsigned short* bb = wswz + ((n_tile * kKsGate + ks) << 9) + laneoff;
      __builtin_prefetch((const void*)(bb + 512), 0, 3);
      v16bf Bf = *(const v16bf*)(bb);
      v16bf Bi = *(const v16bf*)(bb + kGateStride);
      v16bf Bc = *(const v16bf*)(bb + 2 * kGateStride);
      v16bf Bo = *(const v16bf*)(bb + 3 * kGateStride);
      aF0 = WMMA_BF16(A0, Bf, aF0);  aF1 = WMMA_BF16(A1, Bf, aF1);
      aI0 = WMMA_BF16(A0, Bi, aI0);  aI1 = WMMA_BF16(A1, Bi, aI1);
      aC0 = WMMA_BF16(A0, Bc, aC0);  aC1 = WMMA_BF16(A1, Bc, aC1);
      aO0 = WMMA_BF16(A0, Bo, aO0);  aO1 = WMMA_BF16(A1, Bo, aO1);
    }
    for (int ksh = 0; ksh < kKsH; ++ksh) {          // K 512..1535 (h part)
      v16bf A0 = *(const v16bf*)(hp_rd + (ksh * 2 + 0) * 512 + laneoff);
      v16bf A1 = *(const v16bf*)(hp_rd + (ksh * 2 + 1) * 512 + laneoff);
      const unsigned short* bb =
          wswz + ((n_tile * kKsGate + kKsX + ksh) << 9) + laneoff;
      __builtin_prefetch((const void*)(bb + 512), 0, 3);
      v16bf Bf = *(const v16bf*)(bb);
      v16bf Bi = *(const v16bf*)(bb + kGateStride);
      v16bf Bc = *(const v16bf*)(bb + 2 * kGateStride);
      v16bf Bo = *(const v16bf*)(bb + 3 * kGateStride);
      aF0 = WMMA_BF16(A0, Bf, aF0);  aF1 = WMMA_BF16(A1, Bf, aF1);
      aI0 = WMMA_BF16(A0, Bi, aI0);  aI1 = WMMA_BF16(A1, Bi, aI1);
      aC0 = WMMA_BF16(A0, Bc, aC0);  aC1 = WMMA_BF16(A1, Bc, aC1);
      aO0 = WMMA_BF16(A0, Bo, aO0);  aO1 = WMMA_BF16(A1, Bo, aO1);
    }

    // Pointwise state update + scatter h into next-step A-fragment layout.
    // C/D layout: VGPR r, lane<16 -> M=r(+mt*16), lane>=16 -> M=r+8(+mt*16).
    const int ks_h = ncol >> 5;
    const int kk   = ncol & 31;
    const int hi   = (kk >> 3) & 1;
    const int j2   = (kk & 7) | ((kk >> 4) << 3);
    for (int mt = 0; mt < 2; ++mt) {
      v8f vF = mt ? aF1 : aF0;
      v8f vI = mt ? aI1 : aI0;
      v8f vC = mt ? aC1 : aC0;
      v8f vO = mt ? aO1 : aO0;
      for (int r = 0; r < 8; ++r) {
        int M = mt * 16 + r + ((lane >> 4) << 3);
        int idx = M * kDh + ncol;
        float f  = __expf(vF[r]);
        float i  = __expf(vI[r]);
        float ch = tanhf(vC[r]);
        float o  = 1.0f / (1.0f + __expf(-vO[r]));
        float cn = f * c[idx] + i * ch;
        float nn = f * nst[idx] + i;
        c[idx]   = cn;
        nst[idx] = nn;
        float hv = o * (cn / nn);
        h_tail[idx] = hv;                           // fully rewritten each step
        int lane2 = (M & 15) + (hi << 4);
        hp_wr[(ks_h * 2 + mt) * 512 + (lane2 << 4) + j2] = f2bf(hv);
      }
    }
  } else {
    // ---------------- out(t-1) ----------------
    if (t == 0) return;
    const int n_tile = (blockIdx.x - 16) * 4 + wave;  // 0..31
    const int ncol   = n_tile * 16 + (lane & 15);
    v8f a0, a1;
    {
      float bv = bout[ncol];
      for (int r = 0; r < 8; ++r) { a0[r] = bv; a1[r] = bv; }
    }
    for (int ksh = 0; ksh < kKsH; ++ksh) {
      v16bf A0 = *(const v16bf*)(hp_rd + (ksh * 2 + 0) * 512 + laneoff);
      v16bf A1 = *(const v16bf*)(hp_rd + (ksh * 2 + 1) * 512 + laneoff);
      const unsigned short* bb = wout_swz + ((n_tile * kKsH + ksh) << 9) + laneoff;
      __builtin_prefetch((const void*)(bb + 512), 0, 3);
      v16bf Bm = *(const v16bf*)(bb);
      a0 = WMMA_BF16(A0, Bm, a0);
      a1 = WMMA_BF16(A1, Bm, a1);
    }
    for (int mt = 0; mt < 2; ++mt) {
      v8f a = mt ? a1 : a0;
      for (int r = 0; r < 8; ++r) {
        int M = mt * 16 + r + ((lane >> 4) << 3);
        out[(M * kT + (t - 1)) * kDout + ncol] = a[r];
      }
    }
  }
}

// Final output projection for t = T-1 (reads the last h fragments).
__global__ void lstm_out_final(const unsigned short* __restrict__ hp_rd,
                               const unsigned short* __restrict__ wout_swz,
                               const float* __restrict__ bout,
                               float* __restrict__ out) {
  const int tid  = threadIdx.x;
  const int wave = tid >> 5;
  const int lane = tid & 31;
  const int laneoff = lane << 4;
  const int n_tile = blockIdx.x * 4 + wave;         // 0..31
  const int ncol   = n_tile * 16 + (lane & 15);
  v8f a0, a1;
  {
    float bv = bout[ncol];
    for (int r = 0; r < 8; ++r) { a0[r] = bv; a1[r] = bv; }
  }
  for (int ksh = 0; ksh < kKsH; ++ksh) {
    v16bf A0 = *(const v16bf*)(hp_rd + (ksh * 2 + 0) * 512 + laneoff);
    v16bf A1 = *(const v16bf*)(hp_rd + (ksh * 2 + 1) * 512 + laneoff);
    v16bf Bm = *(const v16bf*)(wout_swz + ((n_tile * kKsH + ksh) << 9) + laneoff);
    a0 = WMMA_BF16(A0, Bm, a0);
    a1 = WMMA_BF16(A1, Bm, a1);
  }
  for (int mt = 0; mt < 2; ++mt) {
    v8f a = mt ? a1 : a0;
    for (int r = 0; r < 8; ++r) {
      int M = mt * 16 + r + ((lane >> 4) << 3);
      out[(M * kT + (kT - 1)) * kDout + ncol] = a[r];
    }
  }
}

// ---------------------------------------------------------------------------
// Host. Inputs: x, Wf, bf, Wi, bi, Wc, bc, Wo, bo, Wout, bout (all f32).
// d_out = outputs[32,1024,512] ++ h[32,1024] ++ c[32,1024] ++ n[32,1024].
// ---------------------------------------------------------------------------
extern "C" void kernel_launch(void* const* d_in, const int* in_sizes, int n_in,
                              void* d_out, int out_size, void* d_ws, size_t ws_size,
                              hipStream_t stream) {
  (void)in_sizes; (void)n_in; (void)out_size; (void)ws_size;
  const float* x    = (const float*)d_in[0];
  const float* Wf   = (const float*)d_in[1];
  const float* bf_  = (const float*)d_in[2];
  const float* Wi   = (const float*)d_in[3];
  const float* bi_  = (const float*)d_in[4];
  const float* Wc   = (const float*)d_in[5];
  const float* bc_  = (const float*)d_in[6];
  const float* Wo   = (const float*)d_in[7];
  const float* bo_  = (const float*)d_in[8];
  const float* Wout = (const float*)d_in[9];
  const float* bout = (const float*)d_in[10];

  float* out    = (float*)d_out;
  float* h_tail = out + (size_t)kB * kT * kDout;    // tuple tail: h (rewritten every step)
  float* c      = h_tail + kB * kDh;                // c
  float* nst    = c + kB * kDh;                     // n

  unsigned short* wswz     = (unsigned short*)d_ws;         // 12.6 MB bf16
  unsigned short* wout_swz = wswz + kGateSwzHalfs;          // 1.0 MB
  unsigned short* xpack    = wout_swz + kOutSwzHalfs;       // 33.5 MB
  unsigned short* hpack0   = xpack + (size_t)kT * kXPackPerT;
  unsigned short* hpack1   = hpack0 + kHPackHalfs;          // 64 KB each

  // State init (idempotent, every call): c,n zero; hpack0 zero (h_{-1}=0).
  zero_f32<<<256, 256, 0, stream>>>(c, 2 * kB * kDh);
  zero_f32<<<64, 256, 0, stream>>>((float*)hpack0, kHPackHalfs / 2);

  // One-time transforms (L2-resident afterwards).
  swizzle_weights<<<(kGateSwzHalfs + kOutSwzHalfs) / 256, 256, 0, stream>>>(
      Wf, Wi, Wc, Wo, Wout, wswz);
  pack_x<<<(kT * kXPackPerT) / 256, 256, 0, stream>>>(x, xpack);

  // Sequential scan; one fused kernel per step (gates(t) + out(t-1)).
  for (int t = 0; t < kT; ++t) {
    const unsigned short* rd = (t & 1) ? hpack1 : hpack0;
    unsigned short*       wr = (t & 1) ? hpack0 : hpack1;
    lstm_step<<<24, 128, 0, stream>>>(xpack, rd, wr, wswz, wout_swz,
                                      bf_, bi_, bc_, bo_, bout,
                                      h_tail, c, nst, out, t);
  }
  // out for t = T-1: h_{T-1} frags were written to hpack[(1023+1)&1] = hpack0.
  lstm_out_final<<<8, 128, 0, stream>>>(hpack0, wout_swz, bout, out);
}
